// TransformerSampling_80384607912547
// MI455X (gfx1250) — matrix operand order repaired and checked by reading, not compiled, not run on hardware
//
#include <hip/hip_runtime.h>
#include <hip/hip_bf16.h>
#include <math.h>
#include <stdint.h>

typedef __attribute__((ext_vector_type(16))) _Float16 v16h;
typedef __attribute__((ext_vector_type(8)))  _Float16 v8h;
typedef __attribute__((ext_vector_type(8)))  float    v8f;
typedef __attribute__((ext_vector_type(4)))  unsigned tdm_v4u;
typedef __attribute__((ext_vector_type(8)))  int      tdm_v8i;
typedef __attribute__((ext_vector_type(4)))  int      tdm_v4i;

#define DIM   192
#define HEADS 6
#define HD    32
#define HID   768
#define WS7   7
#define WU14  14
#define NTOK_BIG   100352   // 2*224*224
#define NTOK_SMALL 25088    // 2*112*112
#define NWIN       512      // 2 * 16*16
#define NMT   13            // M-tiles of 16 covering 196 rows
#define BIASC_ELEMS (HEADS * NMT * 4 * 32 * 8)

#if __has_builtin(__builtin_amdgcn_tensor_load_to_lds)
#define USE_TDM 1
#else
#define USE_TDM 0
#endif

static __device__ __forceinline__ v16h cat16(v8h lo, v8h hi) {
    v16h r;
#pragma unroll
    for (int e = 0; e < 8; ++e) { r[e] = lo[e]; r[e + 8] = hi[e]; }
    return r;
}

static __device__ __forceinline__ float gelu_exact(float x) {
    return 0.5f * x * (1.0f + erff(x * 0.70710678118654752f));
}

// ------------------- cross-lane reductions via v_permlane16 (VALU, no LDS) ---
#if __has_builtin(__builtin_amdgcn_permlane16) && __has_builtin(__builtin_amdgcn_permlanex16)
static __device__ __forceinline__ float plx(float v, unsigned s0, unsigned s1) {
    return __int_as_float(__builtin_amdgcn_permlane16(
        __float_as_int(v), __float_as_int(v), (int)s0, (int)s1, false, false));
}
static __device__ __forceinline__ float plswap16(float v) {   // lane ^= 16
    return __int_as_float(__builtin_amdgcn_permlanex16(
        __float_as_int(v), __float_as_int(v), (int)0x76543210, (int)0xFEDCBA98,
        false, false));
}
static __device__ __forceinline__ float red16_max(float x) {  // max over 16-lane half
    x = fmaxf(x, plx(x, 0x67452301u, 0xEFCDAB89u));   // xor 1
    x = fmaxf(x, plx(x, 0x54761032u, 0xDCFE98BAu));   // xor 2
    x = fmaxf(x, plx(x, 0x32107654u, 0xBA98FEDCu));   // xor 4
    x = fmaxf(x, plx(x, 0xFEDCBA98u, 0x76543210u));   // xor 8
    return x;
}
static __device__ __forceinline__ float red16_sum(float x) {
    x += plx(x, 0x67452301u, 0xEFCDAB89u);
    x += plx(x, 0x54761032u, 0xDCFE98BAu);
    x += plx(x, 0x32107654u, 0xBA98FEDCu);
    x += plx(x, 0xFEDCBA98u, 0x76543210u);
    return x;
}
static __device__ __forceinline__ float red32_sum(float x) {
    x = red16_sum(x);
    x += plswap16(x);
    return x;
}
#else
static __device__ __forceinline__ float red16_max(float x) {
    for (int d = 1; d < 16; d <<= 1) x = fmaxf(x, __shfl_xor(x, d, 32));
    return x;
}
static __device__ __forceinline__ float red16_sum(float x) {
    for (int d = 1; d < 16; d <<= 1) x += __shfl_xor(x, d, 32);
    return x;
}
static __device__ __forceinline__ float red32_sum(float x) {
    for (int d = 1; d < 32; d <<= 1) x += __shfl_xor(x, d, 32);
    return x;
}
#endif

#if USE_TDM
// Tensor Data Mover: load a (rows x cols) f16 tile (row-major, row stride
// `stride` elements) from global into LDS at byte offset ldsOff.
static __device__ __forceinline__ void tdm_load_tile(unsigned ldsOff,
                                                     const _Float16* gptr,
                                                     int rows, int cols, int stride) {
    unsigned long long ga = (unsigned long long)(uintptr_t)gptr;
    tdm_v4u g0;
    g0[0] = 1u;                                           // count=1, user descriptor
    g0[1] = ldsOff;                                       // lds_addr (bytes)
    g0[2] = (unsigned)(ga & 0xffffffffu);                 // global_addr[31:0]
    g0[3] = (unsigned)((ga >> 32) & 0x01ffffffu) | (2u << 30);  // addr[56:32] | type=2
    unsigned td0 = (unsigned)cols, td1 = (unsigned)rows;
    tdm_v8i g1;
    g1[0] = (int)(1u << 16);                              // data_size=1 -> 2 bytes
    g1[1] = (int)((td0 & 0xffffu) << 16);                 // tensor_dim0[15:0]
    g1[2] = (int)(((td0 >> 16) & 0xffffu) | ((td1 & 0xffffu) << 16));
    g1[3] = (int)(((td1 >> 16) & 0xffffu) | ((unsigned)cols << 16));  // tile_dim0
    g1[4] = (int)rows;                                    // tile_dim1 (tile_dim2=0)
    g1[5] = (int)stride;                                  // tensor_dim0_stride[31:0]
    g1[6] = 0;
    g1[7] = 0;
    tdm_v4i z4 = {0, 0, 0, 0};
#if defined(__clang_major__) && (__clang_major__ >= 23)
    tdm_v8i z8 = {0, 0, 0, 0, 0, 0, 0, 0};
    __builtin_amdgcn_tensor_load_to_lds(g0, g1, z4, z4, z8, 0);
#else
    __builtin_amdgcn_tensor_load_to_lds(g0, g1, z4, z4, 0);
#endif
}
#endif

// ------------------------------------------- f32 (KxN) -> f16 transposed (NxK)
__global__ __launch_bounds__(256) void cvt_wT_kernel(const float* __restrict__ in,
                                                     _Float16* __restrict__ out,
                                                     int K, int N) {
    int i = blockIdx.x * 256 + threadIdx.x;
    if (i < K * N) {
        int n = i / K, k = i - n * K;
        out[i] = (_Float16)in[(size_t)k * N + n];
    }
}

// -------------------- masked rel-pos bias, pre-arranged in WMMA C-frag layout
__global__ __launch_bounds__(256) void bias_prep_kernel(const float* __restrict__ rpb,
                                                        float* __restrict__ biasC) {
    int i = blockIdx.x * 256 + threadIdx.x;
    if (i >= BIASC_ELEMS) return;
    int v = i & 7;
    int lane = (i >> 3) & 31;
    int j = (i >> 8) & 3;
    int rest = i >> 10;
    int mt = rest % NMT, head = rest / NMT;
    int row = mt * 16 + (lane >> 4) * 8 + v;
    int col = j * 16 + (lane & 15);
    float out = -1e30f;
    if (row < 196 && col < 49) {
        int bi = row / WU14, bj = row - bi * WU14;
        int si = col / WS7, sj = col - si * WS7;
        out = rpb[((bi - si + 13) * 27 + (bj - sj + 13)) * HEADS + head];
    }
    biasC[i] = out;
}

// --------------------------------------------- LayerNorm + window partition
__global__ __launch_bounds__(256) void ln_part_kernel(const float* __restrict__ x,
                                                      const float* __restrict__ w,
                                                      const float* __restrict__ b,
                                                      _Float16* __restrict__ out,
                                                      int Himg, int ws) {
    int tok  = blockIdx.x * 8 + (threadIdx.x >> 5);
    int lane = threadIdx.x & 31;
    int HW = Himg * Himg;
    int bImg = tok / HW, rem = tok % HW;
    int h = rem / Himg, wi = rem % Himg;
    const float* base = x + (size_t)bImg * DIM * HW + (size_t)h * Himg + wi;

    float vv[6]; float s = 0.f;
#pragma unroll
    for (int j = 0; j < 6; ++j) { vv[j] = base[(size_t)(lane + j * 32) * HW]; s += vv[j]; }
    s = red32_sum(s);
    float mu = s * (1.f / DIM);
    float vs = 0.f;
#pragma unroll
    for (int j = 0; j < 6; ++j) { float t = vv[j] - mu; vs += t * t; }
    vs = red32_sum(vs);
    float inv = rsqrtf(vs * (1.f / DIM) + 1e-5f);

    int nwin = Himg / ws;
    int win = (bImg * nwin + h / ws) * nwin + wi / ws;
    int local = (h % ws) * ws + (wi % ws);
    _Float16* o = out + ((size_t)win * (ws * ws) + local) * DIM;
#pragma unroll
    for (int j = 0; j < 6; ++j) {
        int c = lane + j * 32;
        o[c] = (_Float16)((vv[j] - mu) * inv * w[c] + b[c]);
    }
}

// ---------------------------------------------------- LayerNorm (token-major)
__global__ __launch_bounds__(256) void ln_rows_kernel(const float* __restrict__ x,
                                                      const float* __restrict__ w,
                                                      const float* __restrict__ b,
                                                      _Float16* __restrict__ out) {
    int tok  = blockIdx.x * 8 + (threadIdx.x >> 5);
    int lane = threadIdx.x & 31;
    const float* base = x + (size_t)tok * DIM;
    float vv[6]; float s = 0.f;
#pragma unroll
    for (int j = 0; j < 6; ++j) { vv[j] = base[lane + j * 32]; s += vv[j]; }
    s = red32_sum(s);
    float mu = s * (1.f / DIM);
    float vs = 0.f;
#pragma unroll
    for (int j = 0; j < 6; ++j) { float t = vv[j] - mu; vs += t * t; }
    vs = red32_sum(vs);
    float inv = rsqrtf(vs * (1.f / DIM) + 1e-5f);
    _Float16* o = out + (size_t)tok * DIM;
#pragma unroll
    for (int j = 0; j < 6; ++j) {
        int c = lane + j * 32;
        o[c] = (_Float16)((vv[j] - mu) * inv * w[c] + b[c]);
    }
}

// --------------------------------------------------------------- tiled GEMM
// C(MxN) = A(MxK,f16) * Bt(NxK,f16 pre-transposed) + bias
// 128x64 tile, 4 waves, WMMA f16->f32; double-buffered TDM tile pipeline.
enum { EP_NONE = 0, EP_GELU = 1, EP_PROJ = 2, EP_FC2 = 3 };

template <int EPILOG>
__global__ __launch_bounds__(128) void gemm_kernel(
    const _Float16* __restrict__ A, int lda,
    const _Float16* __restrict__ Bt, int ldb,   // Bt: N rows x K cols
    const float* __restrict__ bias,
    int M, int N, int K,
    _Float16* __restrict__ outh,
    const float* __restrict__ resid,            // EP_PROJ: big_x NCHW ; EP_FC2: xbuf
    float* __restrict__ outf)                   // EP_PROJ: xbuf ; EP_FC2: d_out
{
#if USE_TDM
    __shared__ _Float16 sA[2][128][32];
    __shared__ _Float16 sBt[2][64][32];
#else
    __shared__ _Float16 sA[1][128][32];
    __shared__ _Float16 sBt[1][64][32];
#endif

    const int tid = threadIdx.x;
    const int wave = tid >> 5, lane = tid & 31;
    const int half = lane >> 4, l16 = lane & 15;
    const int m0 = blockIdx.x * 128;
    const int n0 = blockIdx.y * 64;

    v8f acc[2][4];
    const v8f vzero = {0.f, 0.f, 0.f, 0.f, 0.f, 0.f, 0.f, 0.f};
#pragma unroll
    for (int i = 0; i < 2; ++i)
#pragma unroll
        for (int j = 0; j < 4; ++j) acc[i][j] = vzero;

    const int nk = K >> 5;
#if USE_TDM
    if (tid < 32) {        // prologue: DMA first tiles; TDM ignores EXEC
        tdm_load_tile((unsigned)(uintptr_t)&sA[0][0][0],
                      A + (size_t)m0 * lda, 128, 32, lda);
        tdm_load_tile((unsigned)(uintptr_t)&sBt[0][0][0],
                      Bt + (size_t)n0 * ldb, 64, 32, ldb);
    }
#endif
    for (int kk = 0; kk < nk; ++kk) {
        const int cur = kk & 1;
#if USE_TDM
        if (tid < 32) __builtin_amdgcn_s_wait_tensorcnt(0);
        __syncthreads();
        if (kk + 1 < nk && tid < 32) {     // overlap next DMA with this compute
            int k1 = (kk + 1) << 5;
            tdm_load_tile((unsigned)(uintptr_t)&sA[cur ^ 1][0][0],
                          A + (size_t)m0 * lda + k1, 128, 32, lda);
            tdm_load_tile((unsigned)(uintptr_t)&sBt[cur ^ 1][0][0],
                          Bt + (size_t)n0 * ldb + k1, 64, 32, ldb);
        }
#else
        int k0 = kk << 5;
#pragma unroll
        for (int i = 0; i < 4; ++i) {
            int c = tid + i * 128;
            int r = c >> 2, off = (c & 3) * 8;
            *(v8h*)&sA[0][r][off] = *(const v8h*)(A + (size_t)(m0 + r) * lda + k0 + off);
        }
#pragma unroll
        for (int i = 0; i < 2; ++i) {
            int c = tid + i * 128;
            int n = c >> 2, off = (c & 3) * 8;
            *(v8h*)&sBt[0][n][off] = *(const v8h*)(Bt + (size_t)(n0 + n) * ldb + k0 + off);
        }
        __syncthreads();
#endif

        v16h aF[2], bF[4];
#pragma unroll
        for (int i = 0; i < 2; ++i) {
            int row = wave * 32 + i * 16 + l16;
            v8h lo = *(const v8h*)&sA[cur][row][half * 8];
            v8h hi = *(const v8h*)&sA[cur][row][16 + half * 8];
            aF[i] = cat16(lo, hi);
        }
#pragma unroll
        for (int j = 0; j < 4; ++j) {
            int col = j * 16 + l16;
            v8h lo = *(const v8h*)&sBt[cur][col][half * 8];
            v8h hi = *(const v8h*)&sBt[cur][col][16 + half * 8];
            bF[j] = cat16(lo, hi);
        }
#pragma unroll
        for (int i = 0; i < 2; ++i)
#pragma unroll
            for (int j = 0; j < 4; ++j)
                acc[i][j] = __builtin_amdgcn_wmma_f32_16x16x32_f16(
                    false, aF[i], false, bF[j], (short)0, acc[i][j], false, false);
#if !USE_TDM
        __syncthreads();
#endif
    }

    // epilogue (C layout: row-in-tile = v + 8*half, col-in-tile = l16)
#pragma unroll
    for (int i = 0; i < 2; ++i)
#pragma unroll
        for (int j = 0; j < 4; ++j)
#pragma unroll
            for (int v = 0; v < 8; ++v) {
                int row = m0 + wave * 32 + i * 16 + half * 8 + v;
                int col = n0 + j * 16 + l16;
                float val = acc[i][j][v] + bias[col];
                if (EPILOG == EP_NONE) {
                    outh[(size_t)row * N + col] = (_Float16)val;
                } else if (EPILOG == EP_GELU) {
                    outh[(size_t)row * N + col] = (_Float16)gelu_exact(val);
                } else if (EPILOG == EP_PROJ) {
                    int win = row / 196, local = row % 196;
                    int b = win >> 8, widx = win & 255;
                    int h = (widx >> 4) * WU14 + local / WU14;
                    int w = (widx & 15) * WU14 + local % WU14;
                    float sc = resid[(((size_t)b * DIM + col) * 224 + h) * 224 + w];
                    outf[((size_t)(b * 50176 + h * 224 + w)) * DIM + col] = sc + val;
                } else {  // EP_FC2
                    float r = resid[(size_t)row * DIM + col] + val;
                    int b = row / 50176, rem = row % 50176;
                    int h = rem / 224, w = rem % 224;
                    outf[(((size_t)b * DIM + col) * 224 + h) * 224 + w] = r;
                }
            }
}

// ------------------------------------------------------------- attention
// one block per (window, head); 8 waves, each owns M-tiles of 16 rows.
// S = Q*K^T + biasC (bias preloaded into the WMMA accumulator, mask = -1e30)
__global__ __launch_bounds__(256) void attn_kernel(const _Float16* __restrict__ Q,
                                                   const _Float16* __restrict__ KV,
                                                   const float* __restrict__ biasC,
                                                   _Float16* __restrict__ O) {
    __shared__ _Float16 sK[64][32];     // [small tok][dim], zero padded 49->64
    __shared__ _Float16 sVt[32][64];    // [dim][small tok], zero padded
    __shared__ _Float16 sP[8][16][64];

    const int win = blockIdx.x, head = blockIdx.y;
    const int tid = threadIdx.x, wave = tid >> 5, lane = tid & 31;
    const int half = lane >> 4, l16 = lane & 15;
    const v8h hzero = {};

    const _Float16* Kb = KV + (size_t)win * 49 * 384 + head * HD;
    const _Float16* Vb = KV + (size_t)win * 49 * 384 + DIM + head * HD;
    const _Float16* Qb = Q + (size_t)win * 196 * DIM + head * HD;
    const float* bC = biasC + ((size_t)head * NMT) * 1024 + lane * 8;

    {   // stage K (vector, zero-padded), V transposed (scalar scatter)
        int r = tid >> 2, off = (tid & 3) * 8;      // 256 chunks of 8 f16
        v8h kv = hzero;
        if (r < 49) kv = *(const v8h*)(Kb + (size_t)r * 384 + off);
        *(v8h*)&sK[r][off] = kv;
        if (r < 49) {
            v8h vv = *(const v8h*)(Vb + (size_t)r * 384 + off);
#pragma unroll
            for (int e = 0; e < 8; ++e) sVt[off + e][r] = vv[e];
        } else {
#pragma unroll
            for (int e = 0; e < 8; ++e) sVt[off + e][r] = (_Float16)0.f;
        }
    }
    __syncthreads();

    // K^T B-fragments (4 N-tiles over small tokens, K-dim = 32 features)
    v16h kF[4];
#pragma unroll
    for (int j = 0; j < 4; ++j) {
        v8h lo = *(const v8h*)&sK[j * 16 + l16][half * 8];
        v8h hi = *(const v8h*)&sK[j * 16 + l16][16 + half * 8];
        kF[j] = cat16(lo, hi);
    }
    // V B-fragments (K-dim = small token padded to 64, N = head dim 32)
    v16h vF[2][2];
#pragma unroll
    for (int kc = 0; kc < 2; ++kc)
#pragma unroll
        for (int nt = 0; nt < 2; ++nt) {
            v8h lo = *(const v8h*)&sVt[nt * 16 + l16][kc * 32 + half * 8];
            v8h hi = *(const v8h*)&sVt[nt * 16 + l16][kc * 32 + 16 + half * 8];
            vF[kc][nt] = cat16(lo, hi);
        }

    const v8f vzero = {0.f, 0.f, 0.f, 0.f, 0.f, 0.f, 0.f, 0.f};
    for (int mt = wave; mt < NMT; mt += 8) {
        int rowBase = mt * 16;
        int qrow = rowBase + l16;
        int qrc = (qrow < 196) ? qrow : 195;        // clamp; rows masked at store
        v8h lo = *(const v8h*)(Qb + (size_t)qrc * DIM + half * 8);
        v8h hi = *(const v8h*)(Qb + (size_t)qrc * DIM + 16 + half * 8);
        v16h qF = cat16(lo, hi);

        v8f s[4];
#pragma unroll
        for (int j = 0; j < 4; ++j) {
            v8f c = *(const v8f*)(bC + (size_t)mt * 1024 + j * 256);
            s[j] = __builtin_amdgcn_wmma_f32_16x16x32_f16(false, qF, false, kF[j],
                                                          (short)0, c, false, false);
        }

        // pure softmax: bias & masking already inside the accumulator
#pragma unroll
        for (int v = 0; v < 8; ++v) {
            float a0 = s[0][v], a1 = s[1][v], a2 = s[2][v], a3 = s[3][v];
            float mx = red16_max(fmaxf(fmaxf(a0, a1), fmaxf(a2, a3)));
            float e0 = __expf(a0 - mx), e1 = __expf(a1 - mx);
            float e2 = __expf(a2 - mx), e3 = __expf(a3 - mx);
            float sum = red16_sum((e0 + e1) + (e2 + e3));
            float inv = 1.0f / sum;
            int r = half * 8 + v;
            sP[wave][r][l16]      = (_Float16)(e0 * inv);
            sP[wave][r][16 + l16] = (_Float16)(e1 * inv);
            sP[wave][r][32 + l16] = (_Float16)(e2 * inv);
            sP[wave][r][48 + l16] = (_Float16)(e3 * inv);
        }

        // O = P(16x64) x V(64x32)
        v8f o[2] = {vzero, vzero};
#pragma unroll
        for (int kc = 0; kc < 2; ++kc) {
            v8h plo = *(const v8h*)&sP[wave][l16][kc * 32 + half * 8];
            v8h phi = *(const v8h*)&sP[wave][l16][kc * 32 + 16 + half * 8];
            v16h pF = cat16(plo, phi);
#pragma unroll
            for (int nt = 0; nt < 2; ++nt)
                o[nt] = __builtin_amdgcn_wmma_f32_16x16x32_f16(false, pF, false, vF[kc][nt],
                                                               (short)0, o[nt], false, false);
        }
#pragma unroll
        for (int nt = 0; nt < 2; ++nt)
#pragma unroll
            for (int v = 0; v < 8; ++v) {
                int row = rowBase + half * 8 + v;
                if (row < 196)
                    O[((size_t)win * 196 + row) * DIM + head * HD + nt * 16 + l16] =
                        (_Float16)o[nt][v];
            }
    }
}

// ------------------------------------------------------------------ launch
extern "C" void kernel_launch(void* const* d_in, const int* in_sizes, int n_in,
                              void* d_out, int out_size, void* d_ws, size_t ws_size,
                              hipStream_t stream) {
    const float* big_x  = (const float*)d_in[0];
    const float* small_x= (const float*)d_in[1];
    const float* norm_w = (const float*)d_in[2];
    const float* norm_b = (const float*)d_in[3];
    const float* qkv_w  = (const float*)d_in[4];
    const float* qkv_b  = (const float*)d_in[5];
    const float* rpb    = (const float*)d_in[6];
    const float* proj_w = (const float*)d_in[7];
    const float* proj_b = (const float*)d_in[8];
    const float* fc1_w  = (const float*)d_in[9];
    const float* fc1_b  = (const float*)d_in[10];
    const float* fc2_w  = (const float*)d_in[11];
    const float* fc2_b  = (const float*)d_in[12];
    float* dout = (float*)d_out;

    char* p = (char*)d_ws;
    auto carve = [&](size_t bytes) -> void* {
        void* r = (void*)p;
        p += (bytes + 255) & ~(size_t)255;
        return r;
    };
    _Float16* wqT    = (_Float16*)carve((size_t)576 * DIM * 2);   // 576 x 192
    _Float16* wprojT = (_Float16*)carve((size_t)DIM * DIM * 2);   // 192 x 192
    _Float16* wfc1T  = (_Float16*)carve((size_t)HID * DIM * 2);   // 768 x 192
    _Float16* wfc2T  = (_Float16*)carve((size_t)DIM * HID * 2);   // 192 x 768
    float*    biasC  = (float*)carve((size_t)BIASC_ELEMS * 4);
    _Float16* bx     = (_Float16*)carve((size_t)NTOK_BIG * DIM * 2);
    _Float16* sx     = (_Float16*)carve((size_t)NTOK_SMALL * DIM * 2);
    _Float16* qbuf   = (_Float16*)carve((size_t)NTOK_BIG * DIM * 2);
    _Float16* kvbuf  = (_Float16*)carve((size_t)NTOK_SMALL * 384 * 2);
    _Float16* ao     = (_Float16*)carve((size_t)NTOK_BIG * DIM * 2);
    float*    xbuf   = (float*)carve((size_t)NTOK_BIG * DIM * 4);
    _Float16* y2     = (_Float16*)carve((size_t)NTOK_BIG * DIM * 2);
    _Float16* h1     = (_Float16*)carve((size_t)NTOK_BIG * HID * 2);

    // weight conversion + transpose (Bt layout: N rows x K cols) + bias prep
    cvt_wT_kernel<<<(DIM * 576 + 255) / 256, 256, 0, stream>>>(qkv_w, wqT, DIM, 576);
    cvt_wT_kernel<<<(DIM * DIM + 255) / 256, 256, 0, stream>>>(proj_w, wprojT, DIM, DIM);
    cvt_wT_kernel<<<(DIM * HID + 255) / 256, 256, 0, stream>>>(fc1_w, wfc1T, DIM, HID);
    cvt_wT_kernel<<<(HID * DIM + 255) / 256, 256, 0, stream>>>(fc2_w, wfc2T, HID, DIM);
    bias_prep_kernel<<<(BIASC_ELEMS + 255) / 256, 256, 0, stream>>>(rpb, biasC);

    // LN + partition
    ln_part_kernel<<<NTOK_BIG / 8, 256, 0, stream>>>(big_x, norm_w, norm_b, bx, 224, WU14);
    ln_part_kernel<<<NTOK_SMALL / 8, 256, 0, stream>>>(small_x, norm_w, norm_b, sx, 112, WS7);

    // Q (big) ; K,V (small)  — Bt row offset selects qkv output columns
    gemm_kernel<EP_NONE><<<dim3(NTOK_BIG / 128, DIM / 64), 128, 0, stream>>>(
        bx, DIM, wqT, DIM, qkv_b, NTOK_BIG, DIM, DIM, qbuf, nullptr, nullptr);
    gemm_kernel<EP_NONE><<<dim3(NTOK_SMALL / 128, 384 / 64), 128, 0, stream>>>(
        sx, DIM, wqT + (size_t)DIM * DIM, DIM, qkv_b + DIM, NTOK_SMALL, 384, DIM,
        kvbuf, nullptr, nullptr);

    // attention
    attn_kernel<<<dim3(NWIN, HEADS), 256, 0, stream>>>(qbuf, kvbuf, biasC, ao);

    // proj + residual (window reverse -> token-major xbuf)
    gemm_kernel<EP_PROJ><<<dim3(NTOK_BIG / 128, DIM / 64), 128, 0, stream>>>(
        ao, DIM, wprojT, DIM, proj_b, NTOK_BIG, DIM, DIM, nullptr, big_x, xbuf);

    // LN2 + MLP + residual (+ NCHW transpose into d_out)
    ln_rows_kernel<<<NTOK_BIG / 8, 256, 0, stream>>>(xbuf, norm_w, norm_b, y2);
    gemm_kernel<EP_GELU><<<dim3(NTOK_BIG / 128, HID / 64), 128, 0, stream>>>(
        y2, DIM, wfc1T, DIM, fc1_b, NTOK_BIG, HID, DIM, h1, nullptr, nullptr);
    gemm_kernel<EP_FC2><<<dim3(NTOK_BIG / 128, DIM / 64), 128, 0, stream>>>(
        h1, HID, wfc2T, HID, fc2_b, NTOK_BIG, DIM, HID, nullptr, xbuf, dout);

    (void)in_sizes; (void)n_in; (void)out_size; (void)ws_size;
}
